// QLSTM_65481071400369
// MI455X (gfx1250) — matrix operand order repaired and checked by reading, not compiled, not run on hardware
//
#include <hip/hip_runtime.h>
#include <hip/hip_bf16.h>
#include <math.h>

// ---------------------------------------------------------------------------
// QLSTM (attention-LSTM scan) for MI455X / gfx1250, wave32 + WMMA bf16.
// S=512 seq, B=256 batch, D=256 in, H=256 hidden, E=D+H=512.
//
// 4 fused kernels per scan step (sequential chain => minimize launches):
//   1) qkv_gemm       : [q|k] + v^T from comb=[x_t|hx]; x pre-converted to
//                       bf16 once per launch (pure-b128 inner loop), with an
//                       on-the-fly f32 fallback if workspace is small.
//   2) scores_softmax : q@k^T -> row softmax -> bf16 probs (LDS-fused)
//   3) attn_gemm      : probs @ v  (v consumed as v^T, contiguous B-frags)
//   4) gates_lstm     : 4-gate GEMM (interleaved weights) + LSTM pointwise
// All GEMMs are C = A(MxK) @ W(NxK)^T so WMMA A/B fragments are contiguous
// 128-bit loads under the CDNA5 VGPR layouts. bf16 in, f32 accumulate.
// ---------------------------------------------------------------------------

#define S_DIM 512
#define B_DIM 256
#define D_DIM 256
#define H_DIM 256
#define E_DIM 512

typedef __attribute__((ext_vector_type(16))) __bf16 v16bf;
typedef __attribute__((ext_vector_type(8)))  float  v8f;

union FragBF { v16bf v; uint4 u[2]; __bf16 h[16]; };

__device__ __forceinline__ __bf16 f2bf(float x) {
  unsigned u = __float_as_uint(x);
  u += 0x7FFFu + ((u >> 16) & 1u);           // round-to-nearest-even
  unsigned short hh = (unsigned short)(u >> 16);
  __bf16 r;
  __builtin_memcpy(&r, &hh, sizeof(r));
  return r;
}

__device__ __forceinline__ float sigmoid_f(float x) {
  return 1.0f / (1.0f + __expf(-x));
}

// A-frag (16x32 bf16): per-lane two contiguous 8-elem K-chunks (p, p+16)
__device__ __forceinline__ void load_abf(const __bf16* p, FragBF& f) {
  f.u[0] = *(const uint4*)(p);
  f.u[1] = *(const uint4*)(p + 16);
}
// B-frag (32x16 bf16): per-lane two contiguous 8-elem K-chunks (p, p+16)
__device__ __forceinline__ void load_bbf(const __bf16* p, FragBF& f) {
  f.u[0] = *(const uint4*)(p);
  f.u[1] = *(const uint4*)(p + 16);
}
// A-frag from f32 source with in-register bf16 conversion (fallback path)
__device__ __forceinline__ void load_af32(const float* p, FragBF& f) {
  float4 f0 = *(const float4*)(p);
  float4 f1 = *(const float4*)(p + 4);
  float4 f2 = *(const float4*)(p + 16);
  float4 f3 = *(const float4*)(p + 20);
  f.h[0] = f2bf(f0.x);  f.h[1] = f2bf(f0.y);  f.h[2]  = f2bf(f0.z);  f.h[3]  = f2bf(f0.w);
  f.h[4] = f2bf(f1.x);  f.h[5] = f2bf(f1.y);  f.h[6]  = f2bf(f1.z);  f.h[7]  = f2bf(f1.w);
  f.h[8] = f2bf(f2.x);  f.h[9] = f2bf(f2.y);  f.h[10] = f2bf(f2.z);  f.h[11] = f2bf(f2.w);
  f.h[12] = f2bf(f3.x); f.h[13] = f2bf(f3.y); f.h[14] = f2bf(f3.z);  f.h[15] = f2bf(f3.w);
}

__device__ __forceinline__ void wmma8(FragBF a[2], FragBF b[4], v8f acc[2][4]) {
#pragma unroll
  for (int i = 0; i < 2; ++i)
#pragma unroll
    for (int j = 0; j < 4; ++j)
      acc[i][j] = __builtin_amdgcn_wmma_f32_16x16x32_bf16(
          false, a[i].v, false, b[j].v, (short)0, acc[i][j], false, false);
}

// Shared QKV epilogue: route [q|k] vs transposed v
__device__ __forceinline__ void qkv_epilogue(
    v8f acc[2][4], int lane, int m0, int n0,
    __bf16* __restrict__ QK, __bf16* __restrict__ VT) {
  const int mrow = (lane >> 4) << 3, ncol = lane & 15;
#pragma unroll
  for (int i = 0; i < 2; ++i)
#pragma unroll
    for (int j = 0; j < 4; ++j)
#pragma unroll
      for (int r = 0; r < 8; ++r) {
        const int m = m0 + 16 * i + mrow + r;
        const int n = n0 + 16 * j + ncol;
        const __bf16 val = f2bf(acc[i][j][r]);
        if (n < 2 * E_DIM) QK[(size_t)m * (2 * E_DIM) + n] = val;          // q|k
        else               VT[(size_t)(n - 2 * E_DIM) * B_DIM + m] = val;  // v^T
      }
}

// ---------------------------------------------------------------------------
// 1a) QKV, primary path: x pre-converted to bf16 -> pure b128 inner loops.
//     grid(6,4), block 256 (8 waves, 2Mx4N, wave tile 32x64).
// ---------------------------------------------------------------------------
__global__ __launch_bounds__(256) void qlstm_qkv_gemm_bf(
    const __bf16* __restrict__ Xs,    // (B x D) bf16, x_t
    const __bf16* __restrict__ Hx,    // (B x H) bf16, h_{t-1}
    const __bf16* __restrict__ Wqkv,  // (1536 x 512) bf16
    __bf16* __restrict__ QK,          // (256 x 1024)
    __bf16* __restrict__ VT)          // (512 x 256) = v^T
{
  const int lane = threadIdx.x & 31, wave = threadIdx.x >> 5;
  const int m0 = blockIdx.y * 64  + (wave >> 2) * 32;
  const int n0 = blockIdx.x * 256 + (wave & 3) * 64;

  const v8f z = {};
  v8f acc[2][4];
#pragma unroll
  for (int i = 0; i < 2; ++i)
#pragma unroll
    for (int j = 0; j < 4; ++j) acc[i][j] = z;

  const int arow = m0 + (lane & 15), akof = (lane >> 4) << 3;
  const int wrow = n0 + (lane & 15), wkof = (lane >> 4) << 3;

  const __bf16* x0 = Xs + (size_t)arow * D_DIM + akof;
  const __bf16* x1 = Xs + (size_t)(arow + 16) * D_DIM + akof;
  const __bf16* h0 = Hx + (size_t)arow * H_DIM + akof;
  const __bf16* h1 = Hx + (size_t)(arow + 16) * H_DIM + akof;
  const __bf16* w0 = Wqkv + (size_t)wrow * E_DIM + wkof;
  const __bf16* w1 = Wqkv + (size_t)(wrow + 16) * E_DIM + wkof;
  const __bf16* w2 = Wqkv + (size_t)(wrow + 32) * E_DIM + wkof;
  const __bf16* w3 = Wqkv + (size_t)(wrow + 48) * E_DIM + wkof;

  FragBF a[2], b[4];
  // phase 1: K = [0, 256) from x_bf
  for (int k0 = 0; k0 < D_DIM; k0 += 32) {
    load_abf(x0 + k0, a[0]);
    load_abf(x1 + k0, a[1]);
    load_bbf(w0 + k0, b[0]);
    load_bbf(w1 + k0, b[1]);
    load_bbf(w2 + k0, b[2]);
    load_bbf(w3 + k0, b[3]);
    wmma8(a, b, acc);
  }
  // phase 2: K = [256, 512) from hx_bf
  for (int k0 = 0; k0 < H_DIM; k0 += 32) {
    load_abf(h0 + k0, a[0]);
    load_abf(h1 + k0, a[1]);
    load_bbf(w0 + D_DIM + k0, b[0]);
    load_bbf(w1 + D_DIM + k0, b[1]);
    load_bbf(w2 + D_DIM + k0, b[2]);
    load_bbf(w3 + D_DIM + k0, b[3]);
    wmma8(a, b, acc);
  }
  qkv_epilogue(acc, lane, m0, n0, QK, VT);
}

// ---------------------------------------------------------------------------
// 1b) QKV, fallback path (small workspace): x converted in-register.
// ---------------------------------------------------------------------------
__global__ __launch_bounds__(256) void qlstm_qkv_gemm_f32x(
    const float* __restrict__ Xs,     // (B x D) f32, x_t
    const __bf16* __restrict__ Hx,    // (B x H) bf16
    const __bf16* __restrict__ Wqkv,  // (1536 x 512) bf16
    __bf16* __restrict__ QK,
    __bf16* __restrict__ VT)
{
  const int lane = threadIdx.x & 31, wave = threadIdx.x >> 5;
  const int m0 = blockIdx.y * 64  + (wave >> 2) * 32;
  const int n0 = blockIdx.x * 256 + (wave & 3) * 64;

  const v8f z = {};
  v8f acc[2][4];
#pragma unroll
  for (int i = 0; i < 2; ++i)
#pragma unroll
    for (int j = 0; j < 4; ++j) acc[i][j] = z;

  const int arow = m0 + (lane & 15), akof = (lane >> 4) << 3;
  const int wrow = n0 + (lane & 15), wkof = (lane >> 4) << 3;

  const float*  a0 = Xs + (size_t)arow * D_DIM + akof;
  const float*  a1 = Xs + (size_t)(arow + 16) * D_DIM + akof;
  const __bf16* h0 = Hx + (size_t)arow * H_DIM + akof;
  const __bf16* h1 = Hx + (size_t)(arow + 16) * H_DIM + akof;
  const __bf16* w0 = Wqkv + (size_t)wrow * E_DIM + wkof;
  const __bf16* w1 = Wqkv + (size_t)(wrow + 16) * E_DIM + wkof;
  const __bf16* w2 = Wqkv + (size_t)(wrow + 32) * E_DIM + wkof;
  const __bf16* w3 = Wqkv + (size_t)(wrow + 48) * E_DIM + wkof;

  FragBF a[2], b[4];
  for (int k0 = 0; k0 < D_DIM; k0 += 32) {
    load_af32(a0 + k0, a[0]);
    load_af32(a1 + k0, a[1]);
    load_bbf(w0 + k0, b[0]);
    load_bbf(w1 + k0, b[1]);
    load_bbf(w2 + k0, b[2]);
    load_bbf(w3 + k0, b[3]);
    wmma8(a, b, acc);
  }
  for (int k0 = 0; k0 < H_DIM; k0 += 32) {
    load_abf(h0 + k0, a[0]);
    load_abf(h1 + k0, a[1]);
    load_bbf(w0 + D_DIM + k0, b[0]);
    load_bbf(w1 + D_DIM + k0, b[1]);
    load_bbf(w2 + D_DIM + k0, b[2]);
    load_bbf(w3 + D_DIM + k0, b[3]);
    wmma8(a, b, acc);
  }
  qkv_epilogue(acc, lane, m0, n0, QK, VT);
}

// ---------------------------------------------------------------------------
// 2) scores+softmax: P = softmax(q @ k^T * scale), bf16 out.
//    grid(1,4): each block owns 64 complete rows (N=256 fits the block tile).
// ---------------------------------------------------------------------------
__global__ __launch_bounds__(256) void qlstm_scores_softmax(
    const __bf16* __restrict__ Q,   // (256 x 512), row stride 1024
    const __bf16* __restrict__ Km,  // (256 x 512), row stride 1024
    __bf16* __restrict__ P,         // (256 x 256)
    float scale)
{
  __shared__ float tile[64 * 257];  // padded: kills 64-bank conflicts
  __shared__ float rmax[64];
  __shared__ float rinv[64];

  const int tid = threadIdx.x;
  const int lane = tid & 31, wave = tid >> 5;
  const int m0 = blockIdx.y * 64 + (wave >> 2) * 32;
  const int n0 = (wave & 3) * 64;

  const v8f z = {};
  v8f acc[2][4];
#pragma unroll
  for (int i = 0; i < 2; ++i)
#pragma unroll
    for (int j = 0; j < 4; ++j) acc[i][j] = z;

  const int arow = m0 + (lane & 15), akof = (lane >> 4) << 3;
  const int wrow = n0 + (lane & 15), wkof = (lane >> 4) << 3;

  FragBF a[2], b[4];
  for (int k0 = 0; k0 < E_DIM; k0 += 32) {
#pragma unroll
    for (int i = 0; i < 2; ++i)
      load_abf(Q + (size_t)(arow + 16 * i) * (2 * E_DIM) + (k0 + akof), a[i]);
#pragma unroll
    for (int j = 0; j < 4; ++j)
      load_bbf(Km + (size_t)(wrow + 16 * j) * (2 * E_DIM) + (k0 + wkof), b[j]);
    wmma8(a, b, acc);
  }

  const int lm0 = (wave >> 2) * 32 + ((lane >> 4) << 3);
  const int ncol = lane & 15;
#pragma unroll
  for (int i = 0; i < 2; ++i)
#pragma unroll
    for (int j = 0; j < 4; ++j)
#pragma unroll
      for (int r = 0; r < 8; ++r)
        tile[(lm0 + 16 * i + r) * 257 + (n0 + 16 * j + ncol)] =
            acc[i][j][r] * scale;
  __syncthreads();

  if (tid < 64) {                       // per-row max and 1/sum(exp)
    const float* row = &tile[tid * 257];
    float mx = -3.402823e38f;
    for (int c = 0; c < B_DIM; ++c) mx = fmaxf(mx, row[c]);
    float s = 0.0f;
    for (int c = 0; c < B_DIM; ++c) s += __expf(row[c] - mx);
    rmax[tid] = mx;
    rinv[tid] = 1.0f / s;
  }
  __syncthreads();

  for (int t = tid; t < 64 * B_DIM; t += 256) {
    const int lr = t >> 8, c = t & (B_DIM - 1);
    const float v = __expf(tile[lr * 257 + c] - rmax[lr]) * rinv[lr];
    P[(size_t)(blockIdx.y * 64 + lr) * B_DIM + c] = f2bf(v);
  }
}

// ---------------------------------------------------------------------------
// 3) attn = probs @ v : A=probs(256x256), W=v^T(512 rows x 256 K) -> bf16
//    grid(2,4).
// ---------------------------------------------------------------------------
__global__ __launch_bounds__(256) void qlstm_attn_gemm(
    const __bf16* __restrict__ Pm,   // (256 x 256)
    const __bf16* __restrict__ VT,   // (512 x 256)
    __bf16* __restrict__ Attn)       // (256 x 512)
{
  const int lane = threadIdx.x & 31, wave = threadIdx.x >> 5;
  const int m0 = blockIdx.y * 64  + (wave >> 2) * 32;
  const int n0 = blockIdx.x * 256 + (wave & 3) * 64;

  const v8f z = {};
  v8f acc[2][4];
#pragma unroll
  for (int i = 0; i < 2; ++i)
#pragma unroll
    for (int j = 0; j < 4; ++j) acc[i][j] = z;

  const int arow = m0 + (lane & 15), akof = (lane >> 4) << 3;
  const int wrow = n0 + (lane & 15), wkof = (lane >> 4) << 3;

  FragBF a[2], b[4];
  for (int k0 = 0; k0 < B_DIM; k0 += 32) {
#pragma unroll
    for (int i = 0; i < 2; ++i)
      load_abf(Pm + (size_t)(arow + 16 * i) * B_DIM + (k0 + akof), a[i]);
#pragma unroll
    for (int j = 0; j < 4; ++j)
      load_bbf(VT + (size_t)(wrow + 16 * j) * B_DIM + (k0 + wkof), b[j]);
    wmma8(a, b, acc);
  }

  const int mrow = (lane >> 4) << 3, ncol = lane & 15;
#pragma unroll
  for (int i = 0; i < 2; ++i)
#pragma unroll
    for (int j = 0; j < 4; ++j)
#pragma unroll
      for (int r = 0; r < 8; ++r) {
        const int m = m0 + 16 * i + mrow + r;
        const int n = n0 + 16 * j + ncol;
        Attn[(size_t)m * E_DIM + n] = f2bf(acc[i][j][r]);
      }
}

// ---------------------------------------------------------------------------
// 4) gates GEMM + LSTM pointwise. Weights interleaved: row (4h+g) of Wg4i is
//    gate g of hidden unit h, so each block's 256-col tile = 64 h * 4 gates.
//    grid(4,4). Updates cx, hx (f32), hx_bf16, and writes stacked output.
// ---------------------------------------------------------------------------
__global__ __launch_bounds__(256) void qlstm_gates_lstm(
    const __bf16* __restrict__ Attn,  // (256 x 512)
    const __bf16* __restrict__ Wg4,   // (1024 x 512), rows interleaved 4h+g
    const float* __restrict__ bfv, const float* __restrict__ biv,
    const float* __restrict__ bgv, const float* __restrict__ bov,
    float* __restrict__ hx, __bf16* __restrict__ hxbf,
    float* __restrict__ cx, float* __restrict__ out_s)
{
  __shared__ float tile[64 * 257];

  const int tid = threadIdx.x;
  const int lane = tid & 31, wave = tid >> 5;
  const int m0 = blockIdx.y * 64  + (wave >> 2) * 32;
  const int n0 = blockIdx.x * 256 + (wave & 3) * 64;

  const v8f z = {};
  v8f acc[2][4];
#pragma unroll
  for (int i = 0; i < 2; ++i)
#pragma unroll
    for (int j = 0; j < 4; ++j) acc[i][j] = z;

  const int arow = m0 + (lane & 15), akof = (lane >> 4) << 3;
  const int wrow = n0 + (lane & 15), wkof = (lane >> 4) << 3;

  FragBF a[2], b[4];
  for (int k0 = 0; k0 < E_DIM; k0 += 32) {
#pragma unroll
    for (int i = 0; i < 2; ++i)
      load_abf(Attn + (size_t)(arow + 16 * i) * E_DIM + (k0 + akof), a[i]);
#pragma unroll
    for (int j = 0; j < 4; ++j)
      load_bbf(Wg4 + (size_t)(wrow + 16 * j) * E_DIM + (k0 + wkof), b[j]);
    wmma8(a, b, acc);
  }

  const int lm0 = (wave >> 2) * 32 + ((lane >> 4) << 3);
  const int ln0 = (wave & 3) * 64;
  const int ncol = lane & 15;
#pragma unroll
  for (int i = 0; i < 2; ++i)
#pragma unroll
    for (int j = 0; j < 4; ++j)
#pragma unroll
      for (int r = 0; r < 8; ++r)
        tile[(lm0 + 16 * i + r) * 257 + (ln0 + 16 * j + ncol)] = acc[i][j][r];
  __syncthreads();

  // 64 rows (b) x 64 hidden units; 4096 pairs / 256 threads = 16 each
  for (int t = tid; t < 64 * 64; t += 256) {
    const int lm = t >> 6, lh = t & 63;
    const int bidx = blockIdx.y * 64 + lm;
    const int h    = blockIdx.x * 64 + lh;
    const float* pr = &tile[lm * 257 + lh * 4];
    const float f = sigmoid_f(pr[0] + bfv[h]);
    const float i = sigmoid_f(pr[1] + biv[h]);
    const float g = tanhf    (pr[2] + bgv[h]);
    const float o = sigmoid_f(pr[3] + bov[h]);
    const int idx = bidx * H_DIM + h;
    const float c  = f * cx[idx] + i * g;
    const float hn = o * tanhf(c);
    cx[idx] = c;
    hx[idx] = hn;
    hxbf[idx] = f2bf(hn);
    out_s[idx] = hn;
  }
}

// ------------------------- prep kernels ------------------------------------
__global__ void qlstm_cvt_bf16(const float* __restrict__ src,
                               __bf16* __restrict__ dst, int n) {
  int i = blockIdx.x * blockDim.x + threadIdx.x;
  if (i < n) dst[i] = f2bf(src[i]);
}

// gate-interleave conversion: dst[(4h+g)*E + e] = src[h*E + e]
__global__ void qlstm_cvt_gate(const float* __restrict__ src,
                               __bf16* __restrict__ dst, int g) {
  int idx = blockIdx.x * blockDim.x + threadIdx.x;   // H*E
  int h = idx >> 9, e = idx & (E_DIM - 1);
  dst[(size_t)((h << 2) | g) * E_DIM + e] = f2bf(src[idx]);
}

__global__ void qlstm_zero_state(float* __restrict__ hx, float* __restrict__ cx,
                                 __bf16* __restrict__ hxbf) {
  int i = blockIdx.x * blockDim.x + threadIdx.x;
  hx[i] = 0.0f; cx[i] = 0.0f; hxbf[i] = f2bf(0.0f);
}

// ---------------------------------------------------------------------------
extern "C" void kernel_launch(void* const* d_in, const int* in_sizes, int n_in,
                              void* d_out, int out_size, void* d_ws, size_t ws_size,
                              hipStream_t stream) {
  (void)in_sizes; (void)n_in; (void)out_size;
  const float* inputs = (const float*)d_in[0];
  const float* Wq  = (const float*)d_in[1];
  const float* Wk  = (const float*)d_in[2];
  const float* Wv  = (const float*)d_in[3];
  const float* Wf  = (const float*)d_in[4];
  const float* bfv = (const float*)d_in[5];
  const float* Wi  = (const float*)d_in[6];
  const float* biv = (const float*)d_in[7];
  const float* Wg  = (const float*)d_in[8];
  const float* bgv = (const float*)d_in[9];
  const float* Wo  = (const float*)d_in[10];
  const float* bov = (const float*)d_in[11];

  // -------- workspace carve (all sizes multiples of 256B) --------
  char* ws = (char*)d_ws;
  size_t off = 0;
  auto carve = [&](size_t bytes) -> char* {
    char* p = ws + off;
    off = (off + bytes + 255) & ~(size_t)255;
    return p;
  };
  __bf16* Wqkv_bf = (__bf16*)carve((size_t)3 * E_DIM * E_DIM * 2); // [Wq;Wk;Wv] 1536x512
  __bf16* Wg4_bf  = (__bf16*)carve((size_t)4 * H_DIM * E_DIM * 2); // interleaved 1024x512
  __bf16* qk_bf   = (__bf16*)carve((size_t)B_DIM * 2 * E_DIM * 2); // 256x1024 [q|k]
  __bf16* vT_bf   = (__bf16*)carve((size_t)E_DIM * B_DIM * 2);     // 512x256
  __bf16* prob_bf = (__bf16*)carve((size_t)B_DIM * B_DIM * 2);     // 256x256
  __bf16* attn_bf = (__bf16*)carve((size_t)B_DIM * E_DIM * 2);     // 256x512
  float*  hx      = (float*) carve((size_t)B_DIM * H_DIM * 4);
  float*  cx      = (float*) carve((size_t)B_DIM * H_DIM * 4);
  __bf16* hx_bf   = (__bf16*)carve((size_t)B_DIM * H_DIM * 2);

  // Primary path: whole input tensor converted to bf16 once (64 MB).
  const size_t x_elems = (size_t)S_DIM * B_DIM * D_DIM;  // 33,554,432
  const bool use_xbf = (ws_size >= off + x_elems * 2 + 256);
  __bf16* x_bf = use_xbf ? (__bf16*)carve(x_elems * 2) : nullptr;

  // -------- one-time per launch: bf16 weights + state + input --------
  const int TPB = 256;
  const int WEE = E_DIM * E_DIM;                   // 262144
  const int WHE = H_DIM * E_DIM;                   // 131072
  qlstm_cvt_bf16<<<WEE / TPB, TPB, 0, stream>>>(Wq, Wqkv_bf, WEE);
  qlstm_cvt_bf16<<<WEE / TPB, TPB, 0, stream>>>(Wk, Wqkv_bf + (size_t)WEE, WEE);
  qlstm_cvt_bf16<<<WEE / TPB, TPB, 0, stream>>>(Wv, Wqkv_bf + (size_t)2 * WEE, WEE);
  qlstm_cvt_gate<<<WHE / TPB, TPB, 0, stream>>>(Wf, Wg4_bf, 0);
  qlstm_cvt_gate<<<WHE / TPB, TPB, 0, stream>>>(Wi, Wg4_bf, 1);
  qlstm_cvt_gate<<<WHE / TPB, TPB, 0, stream>>>(Wg, Wg4_bf, 2);
  qlstm_cvt_gate<<<WHE / TPB, TPB, 0, stream>>>(Wo, Wg4_bf, 3);
  qlstm_zero_state<<<(B_DIM * H_DIM) / TPB, TPB, 0, stream>>>(hx, cx, hx_bf);
  if (use_xbf)
    qlstm_cvt_bf16<<<(int)(x_elems / TPB), TPB, 0, stream>>>(
        inputs, x_bf, (int)x_elems);

  const float scale = 1.0f / sqrtf((float)(D_DIM + H_DIM));
  float* out = (float*)d_out;

  // -------- sequential scan over S: 4 fused launches per step --------
  for (int s = 0; s < S_DIM; ++s) {
    if (use_xbf)
      qlstm_qkv_gemm_bf<<<dim3(6, 4), 256, 0, stream>>>(
          x_bf + (size_t)s * B_DIM * D_DIM, hx_bf, Wqkv_bf, qk_bf, vT_bf);
    else
      qlstm_qkv_gemm_f32x<<<dim3(6, 4), 256, 0, stream>>>(
          inputs + (size_t)s * B_DIM * D_DIM, hx_bf, Wqkv_bf, qk_bf, vT_bf);

    qlstm_scores_softmax<<<dim3(1, 4), 256, 0, stream>>>(
        qk_bf, qk_bf + E_DIM, prob_bf, scale);

    qlstm_attn_gemm<<<dim3(2, 4), 256, 0, stream>>>(prob_bf, vT_bf, attn_bf);

    qlstm_gates_lstm<<<dim3(4, 4), 256, 0, stream>>>(
        attn_bf, Wg4_bf, bfv, biv, bgv, bov,
        hx, hx_bf, cx, out + (size_t)s * B_DIM * H_DIM);
  }

  // outputs: stacked (S,B,H) then hx then cx
  hipMemcpyAsync(out + (size_t)S_DIM * B_DIM * H_DIM, hx,
                 (size_t)B_DIM * H_DIM * sizeof(float),
                 hipMemcpyDeviceToDevice, stream);
  hipMemcpyAsync(out + (size_t)S_DIM * B_DIM * H_DIM + B_DIM * H_DIM, cx,
                 (size_t)B_DIM * H_DIM * sizeof(float),
                 hipMemcpyDeviceToDevice, stream);
}
